// SelfAttnBlock_16011638080182
// MI455X (gfx1250) — matrix operand order repaired and checked by reading, not compiled, not run on hardware
//
#include <hip/hip_runtime.h>
#include <hip/hip_bf16.h>
#include <math.h>

// ---------------------------------------------------------------------------
// SelfAttnBlock for MI455X (gfx1250): BN -> QKV (f16) -> flash attention via
// v_wmma_f32_16x16x32_f16 (transposed-scores formulation: softmax rows live in
// lane-private VGPRs -> 10 cross-lane ops per 32-column step instead of 64+LDS)
// -> output projection + residual.
// ---------------------------------------------------------------------------

typedef __fp16   f16x2 __attribute__((ext_vector_type(2)));   // type returned by cvt_pkrtz
typedef _Float16 h8   __attribute__((ext_vector_type(8)));
typedef _Float16 v16h __attribute__((ext_vector_type(16)));
typedef float    v8f  __attribute__((ext_vector_type(8)));

#define NB 8
#define CH 64
#define HW 4096
#define EPS_BN 1e-5f

union V16H { v16h v; h8 h[2]; int i[8]; };
union H2I  { f16x2 h; int i; };

// A-layout (16x32 f16, M rows, K cols) loader from a row-major [16][ld] f16 tile.
// lane<16: row M=lane, K {kc+0..7, kc+16..23}; lane>=16: row M=lane-16, K {kc+8..15, kc+24..31}
__device__ __forceinline__ v16h load_a16x32(const _Float16* tile, int ld, int kc, int lane) {
    const _Float16* r = tile + (size_t)(lane & 15) * ld;
    int off0 = kc + ((lane >> 4) << 3);
    V16H u;
    u.h[0] = *(const h8*)(r + off0);
    u.h[1] = *(const h8*)(r + off0 + 16);
    return u.v;
}

// B-layout (32x16 f16, K rows, N cols) loader where element (K,N) = tile[N*ld + kc + K].
// lanes 0-15: N=lane, K {0..7, 16..23}; lanes 16-31: N=lane-16, K {8..15, 24..31}
__device__ __forceinline__ v16h load_b32x16(const _Float16* tile, int ld, int kc, int lane) {
    const _Float16* r = tile + (size_t)(lane & 15) * ld + kc + ((lane >> 4) << 3);
    V16H u;
    u.h[0] = *(const h8*)(r);
    u.h[1] = *(const h8*)(r + 16);
    return u.v;
}

// ---------------------------------------------------------------------------
// Kernel 1: per-channel batchnorm stats (mean, rstd)
// ---------------------------------------------------------------------------
__global__ void __launch_bounds__(256) bnstats_kernel(const float* __restrict__ x,
                                                      float* __restrict__ meanv,
                                                      float* __restrict__ rstdv) {
    __shared__ float r1[256];
    __shared__ float r2[256];
    const int c = blockIdx.x;
    const int tid = threadIdx.x;
    float s1 = 0.f, s2 = 0.f;
    for (int idx = tid; idx < NB * HW; idx += 256) {
        int n = idx >> 12;
        int i = idx & (HW - 1);
        float v = x[((size_t)(n * CH + c)) * HW + i];
        s1 += v;
        s2 += v * v;
    }
    r1[tid] = s1; r2[tid] = s2;
    __syncthreads();
    for (int s = 128; s > 0; s >>= 1) {
        if (tid < s) { r1[tid] += r1[tid + s]; r2[tid] += r2[tid + s]; }
        __syncthreads();
    }
    if (tid == 0) {
        float mean = r1[0] * (1.f / 32768.f);
        float var  = r2[0] * (1.f / 32768.f) - mean * mean;
        meanv[c] = mean;
        rstdv[c] = rsqrtf(var + EPS_BN);
    }
}

// ---------------------------------------------------------------------------
// Kernel 2: BN-apply + QKV 1x1 convs; emits f16 in WMMA-friendly layouts:
//   qT, kT : [n][hw][c]   (row = 64 contiguous halves)
//   vB     : [n][c][hw]   (row = 4096 contiguous halves)
// ---------------------------------------------------------------------------
__global__ void __launch_bounds__(256) qkv_kernel(const float* __restrict__ x,
                                                  const float* __restrict__ gamma,
                                                  const float* __restrict__ beta,
                                                  const float* __restrict__ wq, const float* __restrict__ bq,
                                                  const float* __restrict__ wk, const float* __restrict__ bk,
                                                  const float* __restrict__ wv, const float* __restrict__ bv,
                                                  const float* __restrict__ meanv,
                                                  const float* __restrict__ rstdv,
                                                  _Float16* __restrict__ qT,
                                                  _Float16* __restrict__ kT,
                                                  _Float16* __restrict__ vB) {
    __shared__ float xn[64][64];
    __shared__ float wqs[64][64];
    __shared__ float wks[64][64];
    __shared__ float wvs[64][64];
    __shared__ float bqs[64], bks[64], bvs[64];

    const int tid = threadIdx.x;
    const int n  = blockIdx.x >> 6;
    const int p0 = (blockIdx.x & 63) << 6;

    for (int idx = tid; idx < 4096; idx += 256) {
        ((float*)wqs)[idx] = wq[idx];
        ((float*)wks)[idx] = wk[idx];
        ((float*)wvs)[idx] = wv[idx];
    }
    if (tid < 64) { bqs[tid] = bq[tid]; bks[tid] = bk[tid]; bvs[tid] = bv[tid]; }
    for (int idx = tid; idx < 4096; idx += 256) {
        int c = idx >> 6, p = idx & 63;
        float xv = x[((size_t)(n * CH + c)) * HW + p0 + p];
        xn[c][p] = (xv - meanv[c]) * rstdv[c] * gamma[c] + beta[c];
    }
    __syncthreads();

    for (int k = 0; k < 16; ++k) {
        int oidx = tid + (k << 8);
        int o = oidx & 63;          // consecutive threads -> consecutive o (qT/kT coalesce)
        int p = oidx >> 6;
        float aq = bqs[o], ak = bks[o], av = bvs[o];
        #pragma unroll 8
        for (int c = 0; c < 64; ++c) {
            float xv = xn[c][p];
            aq = fmaf(wqs[o][c], xv, aq);
            ak = fmaf(wks[o][c], xv, ak);
            av = fmaf(wvs[o][c], xv, av);
        }
        size_t gp = (size_t)p0 + p;
        qT[((size_t)n * HW + gp) * CH + o] = (_Float16)aq;
        kT[((size_t)n * HW + gp) * CH + o] = (_Float16)ak;
        vB[((size_t)n * CH + o) * HW + gp] = (_Float16)av;
    }
}

// ---------------------------------------------------------------------------
// Kernel 3: fused flash attention, transposed formulation. One wave = one
// 16-col i-tile. Computes S^T[j,i] so each softmax row (fixed i) lives in one
// lane's VGPRs (+ xor-16 partner); O^T[c,i] accumulated with P^T as B operand.
// ---------------------------------------------------------------------------
__global__ void __launch_bounds__(256) attn_kernel(const _Float16* __restrict__ qT,
                                                   const _Float16* __restrict__ kT,
                                                   const _Float16* __restrict__ vB,
                                                   float* __restrict__ res) {
    const int lane = threadIdx.x & 31;
    const int wave = threadIdx.x >> 5;
    const int n    = blockIdx.x >> 5;
    const int iblk = blockIdx.x & 31;
    const int i0   = iblk * 128 + wave * 16;
    const int hi   = lane >> 4;
    const int ln   = lane & 15;

    const _Float16* qTn = qT + (size_t)n * HW * CH;
    const _Float16* kTn = kT + (size_t)n * HW * CH;
    const _Float16* vn  = vB + (size_t)n * CH * HW;

    // Q tile as loop-invariant B operand (K=c split in two 32-chunks)
    const v16h Bq0 = load_b32x16(qTn + (size_t)i0 * CH, CH, 0,  lane);
    const v16h Bq1 = load_b32x16(qTn + (size_t)i0 * CH, CH, 32, lane);

    v8f O0 = {0.f,0.f,0.f,0.f,0.f,0.f,0.f,0.f};
    v8f O1 = O0, O2 = O0, O3 = O0;
    float m = -INFINITY, l = 0.f;

    // preload K A-tiles for j0 = 0 (software pipeline)
    v16h Ak00 = load_a16x32(kTn,                    CH, 0,  lane);
    v16h Ak01 = load_a16x32(kTn,                    CH, 32, lane);
    v16h Ak10 = load_a16x32(kTn + (size_t)16 * CH,  CH, 0,  lane);
    v16h Ak11 = load_a16x32(kTn + (size_t)16 * CH,  CH, 32, lane);

    for (int j0 = 0; j0 < HW; j0 += 32) {
        // V A-tiles for this step; latency covered by the softmax VALU below
        v16h Av0 = load_a16x32(vn + j0,                    HW, 0, lane);
        v16h Av1 = load_a16x32(vn + (size_t)16 * HW + j0,  HW, 0, lane);
        v16h Av2 = load_a16x32(vn + (size_t)32 * HW + j0,  HW, 0, lane);
        v16h Av3 = load_a16x32(vn + (size_t)48 * HW + j0,  HW, 0, lane);

        // scores^T: T0 = rows j0..j0+15, T1 = rows j0+16..j0+31 (cols = i)
        v8f z  = {0.f,0.f,0.f,0.f,0.f,0.f,0.f,0.f};
        v8f T0 = __builtin_amdgcn_wmma_f32_16x16x32_f16(false, Ak00, false, Bq0, (short)0, z,  false, false);
        T0     = __builtin_amdgcn_wmma_f32_16x16x32_f16(false, Ak01, false, Bq1, (short)0, T0, false, false);
        v8f T1 = __builtin_amdgcn_wmma_f32_16x16x32_f16(false, Ak10, false, Bq0, (short)0, z,  false, false);
        T1     = __builtin_amdgcn_wmma_f32_16x16x32_f16(false, Ak11, false, Bq1, (short)0, T1, false, false);

        // pipeline: start loading next step's K tiles now
        const int jn = (j0 + 32 < HW) ? (j0 + 32) : 0;
        Ak00 = load_a16x32(kTn + (size_t)jn * CH,        CH, 0,  lane);
        Ak01 = load_a16x32(kTn + (size_t)jn * CH,        CH, 32, lane);
        Ak10 = load_a16x32(kTn + (size_t)(jn + 16) * CH, CH, 0,  lane);
        Ak11 = load_a16x32(kTn + (size_t)(jn + 16) * CH, CH, 32, lane);
        if (j0 + 64 < HW) {
            __builtin_prefetch(kTn + (size_t)(j0 + 64) * CH, 0, 3);   // near-cache prefetch
            __builtin_prefetch(vn + j0 + 64, 0, 3);
        }

        // ---- online softmax: row (fixed i=ln) = 16 lane-local values + partner
        float s[16];
        #pragma unroll
        for (int v = 0; v < 8; ++v) { s[v] = T0[v] * 0.125f; s[8 + v] = T1[v] * 0.125f; }

        float t[16];
        #pragma unroll
        for (int v = 0; v < 16; ++v) t[v] = s[v];
        #pragma unroll
        for (int st = 8; st >= 1; st >>= 1)
            #pragma unroll
            for (int k = 0; k < 8; ++k) if (k < st) t[k] = fmaxf(t[k], t[k + st]);
        float mx = fmaxf(t[0], __shfl_xor(t[0], 16, 32));   // one cross-lane op
        float nm = fmaxf(m, mx);
        float alpha = __expf(m - nm);
        m = nm;

        float e[16];
        float rs = 0.f;
        #pragma unroll
        for (int v = 0; v < 16; ++v) { e[v] = __expf(s[v] - nm); rs += e[v]; }
        rs += __shfl_xor(rs, 16, 32);                        // one cross-lane op
        l = l * alpha + rs;

        #pragma unroll
        for (int v = 0; v < 8; ++v) {
            O0[v] *= alpha; O1[v] *= alpha; O2[v] *= alpha; O3[v] *= alpha;
        }

        // ---- build P^T as B operand (32x16, K=j, N=i): pack f16 pairs, swap halves
        H2I p0[4], p1[4];
        int q0[4], q1[4];
        #pragma unroll
        for (int r = 0; r < 4; ++r) {
            p0[r].h = __builtin_amdgcn_cvt_pkrtz(e[2 * r],     e[2 * r + 1]);
            p1[r].h = __builtin_amdgcn_cvt_pkrtz(e[8 + 2 * r], e[8 + 2 * r + 1]);
            q0[r] = __shfl_xor(p0[r].i, 16, 32);
            q1[r] = __shfl_xor(p1[r].i, 16, 32);
        }
        V16H bp;
        #pragma unroll
        for (int r = 0; r < 4; ++r) {
            bp.i[r]     = p0[r].i;   // K(lane<16)=j0..7 local / K(lane>=16)=j8..15 local
            bp.i[4 + r] = p1[r].i;   // K=j16..23 local / K=j24..31 local
        }

        // ---- O^T += V (A, M=c, K=j32) x P^T (B), four 16-row c tiles
        O0 = __builtin_amdgcn_wmma_f32_16x16x32_f16(false, Av0, false, bp.v, (short)0, O0, false, false);
        O1 = __builtin_amdgcn_wmma_f32_16x16x32_f16(false, Av1, false, bp.v, (short)0, O1, false, false);
        O2 = __builtin_amdgcn_wmma_f32_16x16x32_f16(false, Av2, false, bp.v, (short)0, O2, false, false);
        O3 = __builtin_amdgcn_wmma_f32_16x16x32_f16(false, Av3, false, bp.v, (short)0, O3, false, false);
    }

    // ---- finalize: O^T tile (M=c=v+8*hi per tile, N=i=ln); 8 consecutive floats per tile
    float inv = __builtin_amdgcn_rcpf(l);          // v_rcp_f32: plenty for prob normalization
    float* resn = res + ((size_t)n * HW + i0 + ln) * CH;
    const int cbase = hi << 3;
    #pragma unroll
    for (int v = 0; v < 8; ++v) {
        resn[ 0 + cbase + v] = O0[v] * inv;
        resn[16 + cbase + v] = O1[v] * inv;
        resn[32 + cbase + v] = O2[v] * inv;
        resn[48 + cbase + v] = O3[v] * inv;
    }
}

// ---------------------------------------------------------------------------
// Kernel 4: y[n,o,h,w] = x + bo[o] + sum_c wo[o,c] * res[n, c*64+h, w]
// (raw reshape [n,hw,c]->[n,c,h,w] folded into the gather of res rows)
// ---------------------------------------------------------------------------
__global__ void __launch_bounds__(256) outproj_kernel(const float* __restrict__ x,
                                                      const float* __restrict__ wo,
                                                      const float* __restrict__ bo,
                                                      const float* __restrict__ res,
                                                      float* __restrict__ out) {
    __shared__ float Ms[64][64];
    __shared__ float wos[64][64];
    const int tid = threadIdx.x;
    const int n = blockIdx.x >> 6;
    const int h = blockIdx.x & 63;

    for (int idx = tid; idx < 4096; idx += 256) {
        int c = idx >> 6, w = idx & 63;
        Ms[c][w] = res[((size_t)(n * HW) + c * 64 + h) * CH + w];
        ((float*)wos)[idx] = wo[idx];
    }
    __syncthreads();

    for (int k = 0; k < 16; ++k) {
        int oidx = tid + (k << 8);
        int o = oidx >> 6, w = oidx & 63;
        float acc = bo[o];
        #pragma unroll 8
        for (int c = 0; c < 64; ++c) acc = fmaf(wos[o][c], Ms[c][w], acc);
        size_t gi = ((size_t)(n * CH + o)) * HW + h * 64 + w;
        out[gi] = x[gi] + acc;
    }
}

// ---------------------------------------------------------------------------
extern "C" void kernel_launch(void* const* d_in, const int* in_sizes, int n_in,
                              void* d_out, int out_size, void* d_ws, size_t ws_size,
                              hipStream_t stream) {
    (void)in_sizes; (void)n_in; (void)out_size; (void)ws_size;

    const float* x     = (const float*)d_in[0];
    const float* gamma = (const float*)d_in[1];
    const float* beta  = (const float*)d_in[2];
    const float* wq    = (const float*)d_in[3];
    const float* bq    = (const float*)d_in[4];
    const float* wk    = (const float*)d_in[5];
    const float* bk    = (const float*)d_in[6];
    const float* wv    = (const float*)d_in[7];
    const float* bv    = (const float*)d_in[8];
    const float* wo    = (const float*)d_in[9];
    const float* bo    = (const float*)d_in[10];

    char* ws = (char*)d_ws;
    float*    meanv = (float*)ws;                                   // 64 f32
    float*    rstdv = meanv + 64;                                   // 64 f32
    const size_t QKV = (size_t)NB * HW * CH;                        // 2M elems
    _Float16* qT = (_Float16*)(ws + 1024);
    _Float16* kT = qT + QKV;
    _Float16* vB = kT + QKV;
    float*    res = (float*)(ws + 1024 + 3 * QKV * sizeof(_Float16));

    bnstats_kernel<<<CH, 256, 0, stream>>>(x, meanv, rstdv);
    qkv_kernel<<<NB * (HW / 64), 256, 0, stream>>>(x, gamma, beta, wq, bq, wk, bk, wv, bv,
                                                   meanv, rstdv, qT, kT, vB);
    attn_kernel<<<NB * (HW / 128), 256, 0, stream>>>(qT, kT, vB, res);
    outproj_kernel<<<NB * 64, 256, 0, stream>>>(x, wo, bo, res, (float*)d_out);
}
